// AttentionBlock_23021024707139
// MI455X (gfx1250) — matrix operand order repaired and checked by reading, not compiled, not run on hardware
//
#include <hip/hip_runtime.h>
#include <hip/hip_bf16.h>
#include <math.h>

// ---------------------------------------------------------------------------
// Problem constants (match reference)
// ---------------------------------------------------------------------------
#define B_     8
#define SEQ    1028          // K_TOTAL
#define D_     768
#define H_     12
#define HD     64            // head dim
#define NREG   4
#define NPATCH 1024
#define DFF    3072
#define MTOK   (B_ * SEQ)    // 8224 token rows
#define MAXPOS 224
#define QUARTER (D_ / 4)     // 192 rope pairs per half

typedef __attribute__((ext_vector_type(16))) _Float16 v16h;
typedef __attribute__((ext_vector_type(8)))  float    v8f;

// Assemble a v16h from 16 contiguous halves (two 16B vector loads).
__device__ __forceinline__ v16h load16(const _Float16* p) {
    union { uint4 u[2]; v16h h; } t;
    t.u[0] = *(const uint4*)p;
    t.u[1] = *(const uint4*)(p + 8);
    return t.h;
}
// A-fragment (16-bit 16x32 layout): per lane-half, halves [lhi*16, +16) (contiguous k).
__device__ __forceinline__ v16h load_a_frag(const _Float16* rowp, int lhi) {
    union { uint4 u[2]; v16h h; } t;
    t.u[0] = *(const uint4*)(rowp + lhi * 16);
    t.u[1] = *(const uint4*)(rowp + lhi * 16 + 8);
    return t.h;
}

// ---------------------------------------------------------------------------
// f32 -> f16 convert
// ---------------------------------------------------------------------------
__global__ void cvt_f32_f16(const float* __restrict__ in, _Float16* __restrict__ out, int n) {
    int i = blockIdx.x * blockDim.x + threadIdx.x;
    if (i < n) out[i] = (_Float16)in[i];
}

// ---------------------------------------------------------------------------
// LayerNorm (+ optional type embedding), f32 in -> f16 out. One block per token.
// is_context is a numpy bool array (1 byte per element).
// ---------------------------------------------------------------------------
template<bool TYPE_EMBED>
__global__ __launch_bounds__(256) void ln_kernel(
    const float* __restrict__ x, const float* __restrict__ w, const float* __restrict__ bb,
    const unsigned char* __restrict__ is_ctx,
    const float* __restrict__ ctx_e, const float* __restrict__ tgt_e,
    _Float16* __restrict__ out)
{
    __shared__ float s1[256], s2[256];
    const int tok = blockIdx.x;
    const float* xr = x + (size_t)tok * D_;
    float a = 0.f, sq = 0.f;
    for (int d = threadIdx.x; d < D_; d += 256) { float v = xr[d]; a += v; sq += v * v; }
    s1[threadIdx.x] = a; s2[threadIdx.x] = sq;
    __syncthreads();
    for (int s = 128; s > 0; s >>= 1) {
        if (threadIdx.x < s) { s1[threadIdx.x] += s1[threadIdx.x + s]; s2[threadIdx.x] += s2[threadIdx.x + s]; }
        __syncthreads();
    }
    const float mu  = s1[0] * (1.0f / D_);
    const float var = s2[0] * (1.0f / D_) - mu * mu;
    const float rs  = rsqrtf(var + 1e-5f);
    bool ctx = false;
    if (TYPE_EMBED) ctx = (is_ctx[tok] != 0);
    for (int d = threadIdx.x; d < D_; d += 256) {
        float v = (xr[d] - mu) * rs * w[d] + bb[d];
        if (TYPE_EMBED) v += ctx ? ctx_e[d] : tgt_e[d];
        out[(size_t)tok * D_ + d] = (_Float16)v;
    }
}

// ---------------------------------------------------------------------------
// Tiled WMMA GEMM: C[M,N] = A[M,Kd](f16) * Bw[Kd,N](f16) + bias (+GELU) (+res)
// Block: 256 threads = 8 waves; tile 128x64; wave (wm, wn) owns 32x32 of C
// (4 WMMAs per K-step). A/B tiles staged with 16B vector stores; the K-major
// transpose for B-fragments happens on the (cheaper) scalar-gather side.
// Next K-tile prefetched via global_prefetch_b8.
// ---------------------------------------------------------------------------
#define BM 128
#define BN 64
#define BK 32
#define APAD 8   // halves
#define BPAD 8

template<bool GELU, bool OUT_F16, bool ADD_RES>
__global__ __launch_bounds__(256) void gemm_wmma(
    const _Float16* __restrict__ A, const _Float16* __restrict__ Bw,
    const float* __restrict__ bias, const float* __restrict__ res,
    float* __restrict__ Cf, _Float16* __restrict__ Ch,
    int M, int Kd, int N)
{
    __shared__ __align__(16) _Float16 As[BM][BK + APAD];        // row-major M x K
    __shared__ __align__(16) _Float16 Bs[BK][BN + BPAD];        // row-major K x N

    const int tid  = threadIdx.x;
    const int wave = tid >> 5, lane = tid & 31;
    const int lrow = lane & 15, lhi = lane >> 4;
    const int wm = wave >> 1, wn = wave & 1;                    // 4 x 2 waves
    const int row0 = blockIdx.y * BM;
    const int col0 = blockIdx.x * BN;

    // staging coordinates (hoisted)
    const int rA = tid >> 1, cA = (tid & 1) << 4;               // A: 16 halves/thread
    const int rB = tid >> 3, cB = (tid & 7) << 3;               // B: 8 halves/thread
    const int grA = row0 + rA;
    const int grAc = grA < M ? grA : (M - 1);                   // clamped for prefetch

    v8f acc[2][2] = {{{}, {}}, {{}, {}}};

    for (int k0 = 0; k0 < Kd; k0 += BK) {
        {   // stage A tile 128x32 (vector)
            uint4 z = make_uint4(0u, 0u, 0u, 0u);
            uint4 a0 = z, a1 = z;
            if (grA < M) {
                const _Float16* src = A + (size_t)grA * Kd + k0 + cA;
                a0 = *(const uint4*)src;
                a1 = *(const uint4*)(src + 8);
            }
            *(uint4*)&As[rA][cA]     = a0;
            *(uint4*)&As[rA][cA + 8] = a1;
        }
        // stage B tile 32x64 row-major (vector)
        *(uint4*)&Bs[rB][cB] = *(const uint4*)(Bw + (size_t)(k0 + rB) * N + col0 + cB);

        if (k0 + BK < Kd) {   // prefetch next K-tile (global_prefetch_b8)
            __builtin_prefetch(A + (size_t)grAc * Kd + k0 + BK + cA, 0, 1);
            __builtin_prefetch(Bw + (size_t)(k0 + BK + rB) * N + col0 + cB, 0, 1);
        }
        __syncthreads();

        v16h af[2], bf[2];
#pragma unroll
        for (int t = 0; t < 2; ++t)
            af[t] = load_a_frag(&As[wm * 32 + t * 16 + lrow][0], lhi);
#pragma unroll
        for (int u = 0; u < 2; ++u) {
            const int n = wn * 32 + u * 16 + lrow;              // this lane's column
#pragma unroll
            for (int i = 0; i < 16; ++i)                        // K = lhi*16+i (scalar gather)
                bf[u][i] = Bs[(lhi << 4) + i][n];
        }

#pragma unroll
        for (int t = 0; t < 2; ++t)
#pragma unroll
            for (int u = 0; u < 2; ++u)
                acc[t][u] = __builtin_amdgcn_wmma_f32_16x16x32_f16(
                    false, af[t], false, bf[u], (short)0, acc[t][u], false, false);
        __syncthreads();
    }

    // Epilogue. C layout: lane (lane&15) holds col; VGPR i = row lhi*8+i.
#pragma unroll
    for (int u = 0; u < 2; ++u) {
        const int col = col0 + wn * 32 + u * 16 + lrow;
        const float bv = bias[col];
#pragma unroll
        for (int t = 0; t < 2; ++t) {
#pragma unroll
            for (int i = 0; i < 8; ++i) {
                int grow = row0 + wm * 32 + t * 16 + lhi * 8 + i;
                if (grow >= M) continue;
                float v = acc[t][u][i] + bv;
                if (GELU) v = 0.5f * v * (1.0f + erff(v * 0.70710678118654752f));
                size_t o = (size_t)grow * N + col;
                if (ADD_RES) v += res[o];
                if (OUT_F16) Ch[o] = (_Float16)v;
                else         Cf[o] = v;
            }
        }
    }
}

// ---------------------------------------------------------------------------
// 2D RoPE on q/k (f16 in-place), patch tokens only. One thread per (b,patch,pair).
// ---------------------------------------------------------------------------
__global__ void rope_kernel(_Float16* __restrict__ qh, _Float16* __restrict__ kh,
                            const int* __restrict__ coords, const float* __restrict__ ropec)
{
    int gid = blockIdx.x * blockDim.x + threadIdx.x;
    const int PAIRS = D_ / 2;  // 384
    if (gid >= B_ * NPATCH * PAIRS) return;
    int j    = gid % PAIRS;
    int rest = gid / PAIRS;
    int p = rest % NPATCH;
    int b = rest / NPATCH;
    int tok = NREG + p;

    int cy = coords[((size_t)b * NPATCH + p) * 2 + 0];
    int cx = coords[((size_t)b * NPATCH + p) * 2 + 1];
    float fy = fminf(fmaxf((float)cy * ((float)(MAXPOS - 1) / (float)MAXPOS), 0.f), (float)(MAXPOS - 1));
    float fx = fminf(fmaxf((float)cx * ((float)(MAXPOS - 1) / (float)MAXPOS), 0.f), (float)(MAXPOS - 1));
    int ypos = (int)fy, xpos = (int)fx;

    int pos, jj;
    if (j < QUARTER) { pos = xpos; jj = j; }           // first half: x rotation
    else             { pos = ypos; jj = j - QUARTER; } // second half: y rotation
    float c = ropec[((size_t)pos * QUARTER + jj) * 2 + 0];
    float s = ropec[((size_t)pos * QUARTER + jj) * 2 + 1];

    size_t off = ((size_t)(b * SEQ + tok)) * D_ + 2 * j;
    float q0 = (float)qh[off], q1 = (float)qh[off + 1];
    qh[off]     = (_Float16)(q0 * c - q1 * s);
    qh[off + 1] = (_Float16)(q0 * s + q1 * c);
    float k0 = (float)kh[off], k1 = (float)kh[off + 1];
    kh[off]     = (_Float16)(k0 * c - k1 * s);
    kh[off + 1] = (_Float16)(k0 * s + k1 * c);
}

// ---------------------------------------------------------------------------
// Flash attention. Block = 4 waves x 16 q-rows; grid (17, B*H).
// K/V tiles (32 x 64) staged once in LDS and shared by all 4 waves.
// Row sums come from a 5th WMMA against an all-ones B operand (no shuffle
// reduction); only the row-max uses ds_bpermute shuffles.
// acc[0..3] = output d-groups, acc[4] = running softmax denominator.
// ---------------------------------------------------------------------------
template<bool TAIL>
__device__ __forceinline__ void attn_step(
    int j0,
    const _Float16* __restrict__ kb, const _Float16* __restrict__ vb,
    const float* const* mrow,
    const v16h& qa0, const v16h& qa1, const v16h& ones,
    float* mstate, v8f* acc,
    _Float16 (*Ks)[72], _Float16 (*Vs)[72], _Float16 (*Pw)[40],
    int tid, int lrow, int lhi)
{
    __syncthreads();   // previous step's LDS reads complete
    {   // stage K,V tiles (32 tokens x 64 dims) with 128 threads, 16 halves each
        const int r = tid >> 2, c = (tid & 3) << 4;
        uint4 z = make_uint4(0u, 0u, 0u, 0u);
        uint4 ka = z, kbv = z, va = z, vbv = z;
        if (!TAIL || (j0 + r) < SEQ) {
            const _Float16* kp = kb + (size_t)(j0 + r) * D_ + c;
            const _Float16* vp = vb + (size_t)(j0 + r) * D_ + c;
            ka  = *(const uint4*)kp;  kbv = *(const uint4*)(kp + 8);
            va  = *(const uint4*)vp;  vbv = *(const uint4*)(vp + 8);
        }
        *(uint4*)&Ks[r][c] = ka;  *(uint4*)&Ks[r][c + 8] = kbv;
        *(uint4*)&Vs[r][c] = va;  *(uint4*)&Vs[r][c + 8] = vbv;
    }
    __syncthreads();

    const float scale = 0.125f;  // 1/sqrt(64)
    float pv[2][8];
#pragma unroll
    for (int jt = 0; jt < 2; ++jt) {
        const _Float16* krow = &Ks[jt * 16 + lrow][0];
        v16h kf0 = load16(krow + lhi * 16);          // d = lhi*16+i      (chunk 0)
        v16h kf1 = load16(krow + 32 + lhi * 16);     // d = 32+lhi*16+i   (chunk 1)
        v8f s = {};
        s = __builtin_amdgcn_wmma_f32_16x16x32_f16(false, qa0, false, kf0, (short)0, s, false, false);
        s = __builtin_amdgcn_wmma_f32_16x16x32_f16(false, qa1, false, kf1, (short)0, s, false, false);
        const int j = j0 + jt * 16 + lrow;
#pragma unroll
        for (int i = 0; i < 8; ++i) {
            if (!TAIL || j < SEQ) pv[jt][i] = s[i] * scale + mrow[i][j];
            else                  pv[jt][i] = -3.0e38f;
        }
    }
    // online-softmax max update (row m lives in one 16-lane half)
#pragma unroll
    for (int i = 0; i < 8; ++i) {
        float rm = fmaxf(pv[0][i], pv[1][i]);
        rm = fmaxf(rm, __shfl_xor(rm, 1, 32));
        rm = fmaxf(rm, __shfl_xor(rm, 2, 32));
        rm = fmaxf(rm, __shfl_xor(rm, 4, 32));
        rm = fmaxf(rm, __shfl_xor(rm, 8, 32));
        float mnew = fmaxf(mstate[i], rm);
        float sf = __expf(mstate[i] - mnew);
        mstate[i] = mnew;
        float p0 = __expf(pv[0][i] - mnew);
        float p1 = __expf(pv[1][i] - mnew);
        acc[0][i] *= sf; acc[1][i] *= sf; acc[2][i] *= sf; acc[3][i] *= sf; acc[4][i] *= sf;
        // stage P (C-layout -> LDS); same-wave DS ops are in-order, no barrier
        Pw[lhi * 8 + i][lrow]      = (_Float16)p0;
        Pw[lhi * 8 + i][16 + lrow] = (_Float16)p1;
    }
    v16h pa = load_a_frag(&Pw[lrow][0], lhi);
    // O += P*V (4 d-groups) and row-sum += P*ones
#pragma unroll
    for (int t = 0; t < 4; ++t) {
        v16h vf;
#pragma unroll
        for (int i = 0; i < 16; ++i)
            vf[i] = Vs[(lhi << 4) + i][t * 16 + lrow];
        acc[t] = __builtin_amdgcn_wmma_f32_16x16x32_f16(false, pa, false, vf, (short)0, acc[t], false, false);
    }
    acc[4] = __builtin_amdgcn_wmma_f32_16x16x32_f16(false, pa, false, ones, (short)0, acc[4], false, false);
}

__global__ __launch_bounds__(128) void attn_kernel(
    const _Float16* __restrict__ qh, const _Float16* __restrict__ kh,
    const _Float16* __restrict__ vh, const float* __restrict__ mask,
    _Float16* __restrict__ outh)
{
    __shared__ __align__(16) _Float16 Ks[32][72];
    __shared__ __align__(16) _Float16 Vs[32][72];
    __shared__ __align__(16) _Float16 P[4][16][40];

    const int bh = blockIdx.y;
    const int b = bh / H_, h = bh % H_;
    const int tid = threadIdx.x;
    const int wave = tid >> 5, lane = tid & 31;
    const int lrow = lane & 15, lhi = lane >> 4;
    const int q0 = blockIdx.x * 64 + wave * 16;

    const _Float16* qb = qh + (size_t)b * SEQ * D_ + (size_t)h * HD;
    const _Float16* kb = kh + (size_t)b * SEQ * D_ + (size_t)h * HD;
    const _Float16* vb = vh + (size_t)b * SEQ * D_ + (size_t)h * HD;
    const float*    mb = mask + (size_t)(b * H_ + h) * SEQ * SEQ;

    // Q A-fragments for the two K=32 chunks of head dim 64
    v16h qa0 = {}, qa1 = {};
    {
        int qr = q0 + lrow;
        if (qr < SEQ) {
            const _Float16* qp = qb + (size_t)qr * D_;
            qa0 = load_a_frag(qp, lhi);
            qa1 = load_a_frag(qp + 32, lhi);
        }
    }
    v16h ones;
#pragma unroll
    for (int i = 0; i < 16; ++i) ones[i] = (_Float16)1.0f;

    // per-row mask pointers (clamped; clamped rows are never stored)
    const float* mrow[8];
#pragma unroll
    for (int i = 0; i < 8; ++i) {
        int qq = q0 + lhi * 8 + i;
        if (qq >= SEQ) qq = SEQ - 1;
        mrow[i] = mb + (size_t)qq * SEQ;
    }

    float mstate[8];
    v8f acc[5] = {{}, {}, {}, {}, {}};
#pragma unroll
    for (int i = 0; i < 8; ++i) mstate[i] = -3.0e38f;

    for (int j0 = 0; j0 < 1024; j0 += 32)
        attn_step<false>(j0, kb, vb, mrow, qa0, qa1, ones, mstate, acc,
                         Ks, Vs, P[wave], tid, lrow, lhi);
    attn_step<true>(1024, kb, vb, mrow, qa0, qa1, ones, mstate, acc,
                    Ks, Vs, P[wave], tid, lrow, lhi);

    // normalize by the WMMA-accumulated row sums and store (f16, feeds Wo GEMM)
#pragma unroll
    for (int i = 0; i < 8; ++i) {
        int qr = q0 + lhi * 8 + i;
        if (qr >= SEQ) continue;
        float inv = 1.0f / acc[4][i];
        _Float16* op = outh + (size_t)(b * SEQ + qr) * D_ + (size_t)h * HD;
#pragma unroll
        for (int t = 0; t < 4; ++t)
            op[t * 16 + lrow] = (_Float16)(acc[t][i] * inv);
    }
}

// ---------------------------------------------------------------------------
// Host orchestration
// ---------------------------------------------------------------------------
extern "C" void kernel_launch(void* const* d_in, const int* in_sizes, int n_in,
                              void* d_out, int out_size, void* d_ws, size_t ws_size,
                              hipStream_t stream) {
    (void)in_sizes; (void)n_in; (void)out_size; (void)ws_size;

    const float* x      = (const float*)d_in[0];
    const float* amask  = (const float*)d_in[1];
    const unsigned char* is_ctx = (const unsigned char*)d_in[2];  // numpy bool = 1 byte
    const int*   coords = (const int*)d_in[3];
    const float* ropec  = (const float*)d_in[4];
    const float* tgt_e  = (const float*)d_in[5];
    const float* ctx_e  = (const float*)d_in[6];
    const float* ln1w   = (const float*)d_in[7];
    const float* ln1b   = (const float*)d_in[8];
    const float* Wq     = (const float*)d_in[9];
    const float* bq     = (const float*)d_in[10];
    const float* Wk     = (const float*)d_in[11];
    const float* bk     = (const float*)d_in[12];
    const float* Wv     = (const float*)d_in[13];
    const float* bv     = (const float*)d_in[14];
    const float* Wo     = (const float*)d_in[15];
    const float* bo     = (const float*)d_in[16];
    const float* ln2w   = (const float*)d_in[17];
    const float* ln2b   = (const float*)d_in[18];
    const float* W1     = (const float*)d_in[19];
    const float* b1     = (const float*)d_in[20];
    const float* W2     = (const float*)d_in[21];
    const float* b2     = (const float*)d_in[22];
    float* out = (float*)d_out;

    char* wsp = (char*)d_ws;
    size_t off = 0;
    auto carve = [&](size_t bytes) -> char* {
        char* p = wsp + off;
        off += (bytes + 255) & ~(size_t)255;
        return p;
    };
    const size_t MD = (size_t)MTOK * D_;
    const size_t MF = (size_t)MTOK * DFF;
    _Float16* xn  = (_Float16*)carve(MD * 2);
    _Float16* Wqh = (_Float16*)carve((size_t)D_ * D_ * 2);
    _Float16* Wkh = (_Float16*)carve((size_t)D_ * D_ * 2);
    _Float16* Wvh = (_Float16*)carve((size_t)D_ * D_ * 2);
    _Float16* Woh = (_Float16*)carve((size_t)D_ * D_ * 2);
    _Float16* W1h = (_Float16*)carve((size_t)D_ * DFF * 2);
    _Float16* W2h = (_Float16*)carve((size_t)DFF * D_ * 2);
    _Float16* qhh = (_Float16*)carve(MD * 2);
    _Float16* khh = (_Float16*)carve(MD * 2);
    _Float16* vhh = (_Float16*)carve(MD * 2);
    _Float16* ah  = (_Float16*)carve(MD * 2);
    float*    x1  = (float*)   carve(MD * 4);
    _Float16* hh  = (_Float16*)carve(MD * 2);
    _Float16* gh  = (_Float16*)carve(MF * 2);

    // 1. LN1 + type embedding -> f16
    ln_kernel<true><<<MTOK, 256, 0, stream>>>(x, ln1w, ln1b, is_ctx, ctx_e, tgt_e, xn);

    // 2. weight converts -> f16
    {
        int nw = D_ * D_;
        int g = (nw + 255) / 256;
        cvt_f32_f16<<<g, 256, 0, stream>>>(Wq, Wqh, nw);
        cvt_f32_f16<<<g, 256, 0, stream>>>(Wk, Wkh, nw);
        cvt_f32_f16<<<g, 256, 0, stream>>>(Wv, Wvh, nw);
        cvt_f32_f16<<<g, 256, 0, stream>>>(Wo, Woh, nw);
        int nf = D_ * DFF;
        int gf = (nf + 255) / 256;
        cvt_f32_f16<<<gf, 256, 0, stream>>>(W1, W1h, nf);
        cvt_f32_f16<<<gf, 256, 0, stream>>>(W2, W2h, nf);
    }

    const dim3 blk(256);
    const dim3 grdD((D_ + BN - 1) / BN, (MTOK + BM - 1) / BM);    // (12, 65)
    const dim3 grdF((DFF + BN - 1) / BN, (MTOK + BM - 1) / BM);   // (48, 65)

    // 3. QKV projections (f16 out)
    gemm_wmma<false, true, false><<<grdD, blk, 0, stream>>>(xn, Wqh, bq, nullptr, nullptr, qhh, MTOK, D_, D_);
    gemm_wmma<false, true, false><<<grdD, blk, 0, stream>>>(xn, Wkh, bk, nullptr, nullptr, khh, MTOK, D_, D_);
    gemm_wmma<false, true, false><<<grdD, blk, 0, stream>>>(xn, Wvh, bv, nullptr, nullptr, vhh, MTOK, D_, D_);

    // 4. 2D RoPE on q/k patch tokens
    {
        int n = B_ * NPATCH * (D_ / 2);
        rope_kernel<<<(n + 255) / 256, 256, 0, stream>>>(qhh, khh, coords, ropec);
    }

    // 5. attention (flash, streams the 405MB mask once)
    attn_kernel<<<dim3((SEQ + 63) / 64, B_ * H_), dim3(128), 0, stream>>>(qhh, khh, vhh, amask, ah);

    // 6. Wo projection + residual -> x1 (f32)
    gemm_wmma<false, false, true><<<grdD, blk, 0, stream>>>(ah, Woh, bo, x, x1, nullptr, MTOK, D_, D_);

    // 7. LN2 -> f16
    ln_kernel<false><<<MTOK, 256, 0, stream>>>(x1, ln2w, ln2b, nullptr, nullptr, nullptr, hh);

    // 8. MLP: W1 + GELU(exact) -> f16, then W2 + residual -> final f32 output
    gemm_wmma<true,  true,  false><<<grdF, blk, 0, stream>>>(hh, W1h, b1, nullptr, nullptr, gh, MTOK, D_, DFF);
    gemm_wmma<false, false, true ><<<grdD, blk, 0, stream>>>(gh, W2h, b2, x1, out, nullptr, MTOK, DFF, D_);
}
